// ContrastiveLoss_64330020159673
// MI455X (gfx1250) — matrix operand order repaired and checked by reading, not compiled, not run on hardware
//
#include <hip/hip_runtime.h>
#include <hip/hip_bf16.h>
#include <math.h>

// ---------------------------------------------------------------------------
// SimCLR NT-Xent loss, fused for MI455X (gfx1250, wave32, WMMA, WGP/LDS).
//   N = 4096, D = 256, temp = 0.5  ->  2N = 8192 rows
// Pipeline:
//   K1: L2-normalize rows, split each fp32 value into f16 hi + f16 lo.
//   K2: exact fp32 positives  pos[i] = <zi_i, zj_i>.
//   K3: fused GEMM + exp row-sum. 8 waves/WG share double-buffered B tiles in
//       LDS (traffic /8); sim = Ahi*Bhi + Ahi*Blo + Alo*Bhi via
//       v_wmma_f32_16x16x32_f16 (fp32-class accuracy from f16 split).
//   K4: deterministic fixed-order reduction -> scalar loss.
// ---------------------------------------------------------------------------

#define NB      4096
#define DIM     256
#define TWO_N   8192
#define NCHUNK  8
#define CHUNK_COLS (TWO_N / NCHUNK)      // 1024 columns per chunk
#define NTILES  (CHUNK_COLS / 16)        // 64 column tiles per chunk
#define WAVES_PER_WG 8
#define INV_TEMP 2.0f                    // 1 / 0.5

typedef __attribute__((ext_vector_type(16))) _Float16 v16h;
typedef __attribute__((ext_vector_type(8)))  _Float16 v8h;
typedef __attribute__((ext_vector_type(8)))  float    v8f;

// ---------------------------------------------------------------------------
// K1: normalize each row of [emb_i; emb_j], store f16 hi/lo split (row-major).
// One wave per row; each lane owns 8 consecutive floats.
__global__ __launch_bounds__(256) void k_normalize_split(
    const float* __restrict__ ei, const float* __restrict__ ej,
    _Float16* __restrict__ rhi, _Float16* __restrict__ rlo)
{
  const int wave = threadIdx.x >> 5;
  const int lane = threadIdx.x & 31;
  const int row  = blockIdx.x * 8 + wave;          // 0 .. 8191 (grid exact)

  const float* src = (row < NB) ? (ei + (size_t)row * DIM)
                                : (ej + (size_t)(row - NB) * DIM);
  float4 x0 = *(const float4*)(src + lane * 8);
  float4 x1 = *(const float4*)(src + lane * 8 + 4);
  float xs[8] = {x0.x, x0.y, x0.z, x0.w, x1.x, x1.y, x1.z, x1.w};

  float s = 0.f;
#pragma unroll
  for (int e = 0; e < 8; ++e) s += xs[e] * xs[e];
#pragma unroll
  for (int m = 16; m >= 1; m >>= 1) s += __shfl_xor(s, m);

  const float inv = 1.0f / sqrtf(s);
  v8h h, l;
#pragma unroll
  for (int e = 0; e < 8; ++e) {
    float z = xs[e] * inv;
    _Float16 zh = (_Float16)z;
    h[e] = zh;
    l[e] = (_Float16)(z - (float)zh);
  }
  *(v8h*)(rhi + (size_t)row * DIM + lane * 8) = h;
  *(v8h*)(rlo + (size_t)row * DIM + lane * 8) = l;
}

// ---------------------------------------------------------------------------
// K2: exact fp32 positives: pos[i] = dot(emb_i[i], emb_j[i]) / (|ei|*|ej|).
// One wave per pair.
__global__ __launch_bounds__(256) void k_positives(
    const float* __restrict__ ei, const float* __restrict__ ej,
    float* __restrict__ pos)
{
  const int wave = threadIdx.x >> 5;
  const int lane = threadIdx.x & 31;
  const int i    = blockIdx.x * 8 + wave;          // 0 .. 4095 (grid exact)

  const float* a = ei + (size_t)i * DIM;
  const float* b = ej + (size_t)i * DIM;
  float d = 0.f, na = 0.f, nb = 0.f;
#pragma unroll
  for (int e = 0; e < 8; ++e) {
    float x = a[lane * 8 + e], y = b[lane * 8 + e];
    d += x * y; na += x * x; nb += y * y;
  }
#pragma unroll
  for (int m = 16; m >= 1; m >>= 1) {
    d  += __shfl_xor(d, m);
    na += __shfl_xor(na, m);
    nb += __shfl_xor(nb, m);
  }
  if (lane == 0) pos[i] = d / (sqrtf(na) * sqrtf(nb));
}

// ---------------------------------------------------------------------------
// K3: fused sim-GEMM + exp row-sum.
// Workgroup = 8 waves; wave w owns row block (blockGroup*8 + w)*16 with its
// A operands (hi+lo) resident in VGPRs. All 8 waves share each 16-column
// B panel (hi+lo, 16x256 f16 each) through double-buffered LDS, so the
// global/L2 B traffic is divided by 8. One barrier per tile.
__global__ __launch_bounds__(256) void k_gemm_expsum(
    const _Float16* __restrict__ rhi, const _Float16* __restrict__ rlo,
    float* __restrict__ partial)
{
  __shared__ _Float16 ldsH[2][16 * DIM];   // 2 x 8 KB
  __shared__ _Float16 ldsL[2][16 * DIM];   // 2 x 8 KB

  const int tid   = threadIdx.x;
  const int wave  = tid >> 5;
  const int lane  = tid & 31;
  const int rg    = blockIdx.x / NCHUNK;          // row group 0..63
  const int ch    = blockIdx.x % NCHUNK;          // column chunk 0..7
  const int rowBase = (rg * WAVES_PER_WG + wave) * 16;
  const int colBase = ch * CHUNK_COLS;
  const bool hig  = lane >= 16;
  const int  l15  = lane & 15;

  // ---- Load A operands: 16x32 f16 tiles, 8 K-blocks, hi and lo planes.
  // Per-lane A layout (ISA 7.12.2): row = l15; low lanes hold K {0..7,16..23},
  // high lanes K {8..15,24..31} as two contiguous 8-half chunks.
  v16h ahi[8], alo[8];
  {
    const _Float16* ph = rhi + (size_t)(rowBase + l15) * DIM;
    const _Float16* pl = rlo + (size_t)(rowBase + l15) * DIM;
    const int sel = hig ? 8 : 0;
#pragma unroll
    for (int kb = 0; kb < 8; ++kb) {
      v8h c0 = *(const v8h*)(ph + kb * 32 + sel);
      v8h c1 = *(const v8h*)(ph + kb * 32 + 16 + sel);
      v8h d0 = *(const v8h*)(pl + kb * 32 + sel);
      v8h d1 = *(const v8h*)(pl + kb * 32 + 16 + sel);
#pragma unroll
      for (int q = 0; q < 8; ++q) {
        ahi[kb][q] = c0[q]; ahi[kb][q + 8] = c1[q];
        alo[kb][q] = d0[q]; alo[kb][q + 8] = d1[q];
      }
    }
  }

  // ---- Stage helper: copy 16-column B panel (hi+lo) into LDS buffer.
  // 256 threads x 16 contiguous halfs each = 4096 halfs per plane.
  auto stage = [&](int t, int buf) {
    const int n0  = colBase + t * 16;
    const int idx = tid * 16;                  // half index in panel
    const int r   = idx >> 8;                  // panel row   (0..15)
    const int c   = idx & 255;                 // panel K col (0,16,..,240)
    const _Float16* gh = rhi + (size_t)(n0 + r) * DIM + c;
    const _Float16* gl = rlo + (size_t)(n0 + r) * DIM + c;
    *(v8h*)(&ldsH[buf][idx])     = *(const v8h*)(gh);
    *(v8h*)(&ldsH[buf][idx + 8]) = *(const v8h*)(gh + 8);
    *(v8h*)(&ldsL[buf][idx])     = *(const v8h*)(gl);
    *(v8h*)(&ldsL[buf][idx + 8]) = *(const v8h*)(gl + 8);
  };

  v8f rowAcc = {};
  const int ksel = hig ? 8 : 0;    // B operand: low lanes K {0..7,16..23}, high {8..15,24..31}

  stage(0, 0);
  __syncthreads();

#pragma unroll 1
  for (int t = 0; t < NTILES; ++t) {
    const int buf = t & 1;
    if (t + 1 < NTILES) stage(t + 1, buf ^ 1);   // overlap copy with WMMAs

    const int n0 = colBase + t * 16;
    v8f acc = {};
#pragma unroll
    for (int kb = 0; kb < 8; ++kb) {
      // Per-lane B layout: column n0 + l15, two 8-half K chunks (same as A).
      const int boff = l15 * DIM + kb * 32 + ksel;
      v8h e0 = *(const v8h*)(&ldsH[buf][boff]);
      v8h e1 = *(const v8h*)(&ldsH[buf][boff + 16]);
      v8h f0 = *(const v8h*)(&ldsL[buf][boff]);
      v8h f1 = *(const v8h*)(&ldsL[buf][boff + 16]);
      v16h bh, bl;
#pragma unroll
      for (int q = 0; q < 8; ++q) {
        bh[q] = e0[q]; bh[q + 8] = e1[q];
        bl[q] = f0[q]; bl[q + 8] = f1[q];
      }
      // sim += Ahi*Bhi + Ahi*Blo + Alo*Bhi   (fp32-accurate f16 split)
      acc = __builtin_amdgcn_wmma_f32_16x16x32_f16(false, ahi[kb], false, bh,
                                                   (short)0, acc, false, false);
      acc = __builtin_amdgcn_wmma_f32_16x16x32_f16(false, ahi[kb], false, bl,
                                                   (short)0, acc, false, false);
      acc = __builtin_amdgcn_wmma_f32_16x16x32_f16(false, alo[kb], false, bh,
                                                   (short)0, acc, false, false);
    }

    // exp(sim/temp), masking the main diagonal (reference: sum - diag).
    // C/D layout: VGPR k, lane l -> row rowBase + k + 8*(l>=16), col n0 + l15.
    const bool diagT = (n0 == rowBase);
#pragma unroll
    for (int k = 0; k < 8; ++k) {
      float e = __expf(INV_TEMP * acc[k]);
      if (diagT && (k + (hig ? 8 : 0)) == l15) e = 0.0f;
      rowAcc[k] += e;
    }

    __syncthreads();   // reads of buf done; stores to buf^1 visible for t+1
  }

  // Reduce each row sum across its 16-lane group (cols), write partials.
#pragma unroll
  for (int k = 0; k < 8; ++k) {
    float v = rowAcc[k];
    v += __shfl_xor(v, 1);
    v += __shfl_xor(v, 2);
    v += __shfl_xor(v, 4);
    v += __shfl_xor(v, 8);
    if (l15 == 0) {
      const int row = rowBase + k + (hig ? 8 : 0);
      partial[(size_t)ch * TWO_N + row] = v;
    }
  }
}

// ---------------------------------------------------------------------------
// K4: deterministic final reduction.
//   loss = (1/2N) * sum_i [ -pos[i%N]/temp + log(sum_c partial[c][i]) ]
__global__ __launch_bounds__(256) void k_finalize(
    const float* __restrict__ partial, const float* __restrict__ pos,
    float* __restrict__ out)
{
  __shared__ float sm[256];
  const int tid = threadIdx.x;
  float local = 0.f;
  for (int i = tid; i < TWO_N; i += 256) {
    float denom = 0.f;
#pragma unroll
    for (int c = 0; c < NCHUNK; ++c) denom += partial[(size_t)c * TWO_N + i];
    float p = pos[i & (NB - 1)];
    local += -INV_TEMP * p + logf(denom);
  }
  sm[tid] = local;
  __syncthreads();
#pragma unroll
  for (int s = 128; s > 0; s >>= 1) {
    if (tid < s) sm[tid] += sm[tid + s];
    __syncthreads();
  }
  if (tid == 0) out[0] = sm[0] * (1.0f / (float)TWO_N);
}

// ---------------------------------------------------------------------------
extern "C" void kernel_launch(void* const* d_in, const int* in_sizes, int n_in,
                              void* d_out, int out_size, void* d_ws, size_t ws_size,
                              hipStream_t stream) {
  const float* ei = (const float*)d_in[0];     // [4096, 256] fp32
  const float* ej = (const float*)d_in[1];     // [4096, 256] fp32
  float* out = (float*)d_out;                  // scalar fp32

  // Workspace layout:
  //   rhi:     8192*256 f16  = 4 MB
  //   rlo:     8192*256 f16  = 4 MB
  //   pos:     4096 f32      = 16 KB
  //   partial: 8*8192 f32    = 256 KB
  char* ws = (char*)d_ws;
  _Float16* rhi = (_Float16*)ws;
  _Float16* rlo = (_Float16*)(ws + (size_t)TWO_N * DIM * sizeof(_Float16));
  float* pos     = (float*)(ws + (size_t)2 * TWO_N * DIM * sizeof(_Float16));
  float* partial = pos + NB;

  k_normalize_split<<<TWO_N / 8, 256, 0, stream>>>(ei, ej, rhi, rlo);
  k_positives<<<NB / 8, 256, 0, stream>>>(ei, ej, pos);
  // 64 row groups (8 waves x 16 rows) x 8 column chunks
  k_gemm_expsum<<<(TWO_N / (16 * WAVES_PER_WG)) * NCHUNK, 256, 0, stream>>>(
      rhi, rlo, partial);
  k_finalize<<<1, 256, 0, stream>>>(partial, pos, out);
}
